// GAT_Model_27401891348550
// MI455X (gfx1250) — matrix operand order repaired and checked by reading, not compiled, not run on hardware
//
#include <hip/hip_runtime.h>
#include <hip/hip_bf16.h>

#define N_NODES 50000
#define N_EDGES 800000
#define F_IN_1  64
#define DD      128
#define NEG_SLOPE 0.01f

typedef __attribute__((ext_vector_type(16))) __bf16 v16bf;
typedef __attribute__((ext_vector_type(8)))  __bf16 v8bf;
typedef __attribute__((ext_vector_type(8)))  float  v8f;

__device__ __forceinline__ __bf16 f32_to_bf16(float f) {
    unsigned int u = __float_as_uint(f);
    u += 0x7FFFu + ((u >> 16) & 1u);             // round-to-nearest-even
    return __builtin_bit_cast(__bf16, (unsigned short)(u >> 16));
}

// ---------------------------------------------------------------------------
// Fused dual GEMM: Z1 = H @ W1^T, Z2 = H @ W2^T  (v_wmma_f32_16x16x32_bf16)
// Block = 256 threads = 8 waves, covers 128 rows x all 128 cols.
// LDS: H tile (128 x K bf16) + both weight matrices (2 x 128 x K bf16).
// Wave w owns column tile w; loops over 8 row tiles, 2 WMMAs per A-fragment.
// ---------------------------------------------------------------------------
__global__ __launch_bounds__(256)
void gemm2_bf16_wmma(const float* __restrict__ H,
                     const float* __restrict__ W1, const float* __restrict__ W2,
                     float* __restrict__ Z1, float* __restrict__ Z2,
                     int n, int K) {
    extern __shared__ __bf16 smem[];
    __bf16* sH  = smem;               // 128 * K
    __bf16* sW1 = sH  + 128 * K;      // DD * K, [d*K + k]
    __bf16* sW2 = sW1 + DD * K;       // DD * K

    const int tid = threadIdx.x;
    const long rowBase = (long)blockIdx.x * 128;

    // stage both weight matrices (bf16) -- amortized over 128 rows
    const int wTot = DD * K;
    for (int i = tid; i < wTot; i += 256) {
        sW1[i] = f32_to_bf16(W1[i]);
        sW2[i] = f32_to_bf16(W2[i]);
    }
    // stage 128 x K tile of H (zero-pad past n)
    const int hTot = 128 * K;
    for (int i = tid; i < hTot; i += 256) {
        int r = i / K, k = i - r * K;
        long row = rowBase + r;
        sH[i] = f32_to_bf16(row < n ? H[row * K + k] : 0.0f);
    }
    __syncthreads();

    const int wave   = tid >> 5;          // column tile 0..7
    const int lane   = tid & 31;
    const int mn     = lane & 15;         // M for A-frag, N for B/C-frag
    const int hiHalf = lane >> 4;
    const int kbA    = hiHalf ? 16 : 0;   // A frag: lanes 0-15 K 0..15, lanes 16-31 K 16..31
    const int kbB    = hiHalf ? 16 : 0;   // B frag: lanes 0-15 K 0..15, lanes 16-31 K 16..31
    const int col    = wave * 16 + mn;
    const __bf16* w1p = sW1 + col * K;
    const __bf16* w2p = sW2 + col * K;

    for (int rt = 0; rt < 8; ++rt) {
        const __bf16* hp = sH + (rt * 16 + mn) * K;
        v8f acc1 = {}, acc2 = {};
        for (int k0 = 0; k0 < K; k0 += 32) {
            v8bf alo = *(const v8bf*)(hp + k0 + kbA);
            v8bf ahi = *(const v8bf*)(hp + k0 + kbA + 8);
            v16bf a = __builtin_shufflevector(alo, ahi,
                        0,1,2,3,4,5,6,7,8,9,10,11,12,13,14,15);
            v8bf b1lo = *(const v8bf*)(w1p + k0 + kbB);
            v8bf b1hi = *(const v8bf*)(w1p + k0 + kbB + 8);
            v16bf b1 = __builtin_shufflevector(b1lo, b1hi,
                        0,1,2,3,4,5,6,7,8,9,10,11,12,13,14,15);
            v8bf b2lo = *(const v8bf*)(w2p + k0 + kbB);
            v8bf b2hi = *(const v8bf*)(w2p + k0 + kbB + 8);
            v16bf b2 = __builtin_shufflevector(b2lo, b2hi,
                        0,1,2,3,4,5,6,7,8,9,10,11,12,13,14,15);
            acc1 = __builtin_amdgcn_wmma_f32_16x16x32_bf16(
                       false, a, false, b1, (short)0, acc1, false, false);
            acc2 = __builtin_amdgcn_wmma_f32_16x16x32_bf16(
                       false, a, false, b2, (short)0, acc2, false, false);
        }
        // C/D layout: VGPR v holds M = v (lanes 0-15) or v+8 (lanes 16-31)
#pragma unroll
        for (int v = 0; v < 8; ++v) {
            long row = rowBase + rt * 16 + (hiHalf ? v + 8 : v);
            if (row < n) {
                Z1[row * DD + col] = acc1[v];
                Z2[row * DD + col] = acc2[v];
            }
        }
    }
}

// ---------------------------------------------------------------------------
// Per-node attention scalars: s[n] = <z[n], w_s>, d[n] = <z[n], w_d>
// ---------------------------------------------------------------------------
__global__ __launch_bounds__(256)
void node_dots(const float* __restrict__ Z, const float* __restrict__ attn_w,
               float* __restrict__ sv, float* __restrict__ dv, int n) {
    int node = blockIdx.x * 8 + (threadIdx.x >> 5);
    int lane = threadIdx.x & 31;
    if (node >= n) return;
    float4 zz = *(const float4*)(Z + (long)node * DD + lane * 4);
    float4 ws = *(const float4*)(attn_w + lane * 4);
    float4 wd = *(const float4*)(attn_w + DD + lane * 4);
    float ps = zz.x * ws.x + zz.y * ws.y + zz.z * ws.z + zz.w * ws.w;
    float pd = zz.x * wd.x + zz.y * wd.y + zz.z * wd.z + zz.w * wd.w;
#pragma unroll
    for (int off = 16; off > 0; off >>= 1) {
        ps += __shfl_down(ps, off);
        pd += __shfl_down(pd, off);
    }
    if (lane == 0) { sv[node] = ps; dv[node] = pd; }
}

// ---------------------------------------------------------------------------
// Per-layer init: m = -inf, denom = 0, z_nb = 0
// ---------------------------------------------------------------------------
__global__ void init_layer(float* __restrict__ m, float* __restrict__ denom,
                           float* __restrict__ znb, int n, int nd) {
    int i = blockIdx.x * blockDim.x + threadIdx.x;
    if (i < n) { m[i] = __uint_as_float(0xFF800000u); denom[i] = 0.0f; }
    if (i < nd) znb[i] = 0.0f;
}

// float atomic max via signed/unsigned int ordering trick
__device__ __forceinline__ void atomicMaxF(float* addr, float v) {
    if (v >= 0.0f) atomicMax((int*)addr, __float_as_int(v));
    else           atomicMin((unsigned int*)addr, __float_as_uint(v));
}

// Compute leaky-relu logit per edge, cache it in ex[], segment-max into m[]
__global__ void edge_max(const int* __restrict__ src, const int* __restrict__ dst,
                         const float* __restrict__ edge_d,
                         const float* __restrict__ sv, const float* __restrict__ dv,
                         const float* __restrict__ fc0_w, const float* __restrict__ attn_w,
                         float* __restrict__ ex, float* __restrict__ m, int e_cnt) {
    int e = blockIdx.x * blockDim.x + threadIdx.x;
    if (e >= e_cnt) return;
    float tscale = fc0_w[0] * attn_w[2 * DD];
    int d = dst[e];
    float a = sv[src[e]] + dv[d] + edge_d[e] * tscale;
    float lr = a > 0.0f ? a : NEG_SLOPE * a;
    ex[e] = lr;
    atomicMaxF(&m[d], lr);
}

// ex[e] <- exp(logit - m[dst]); segment-sum into denom[]
__global__ void edge_expsum(const int* __restrict__ dst, const float* __restrict__ m,
                            float* __restrict__ ex, float* __restrict__ denom, int e_cnt) {
    int e = blockIdx.x * blockDim.x + threadIdx.x;
    if (e >= e_cnt) return;
    int d = dst[e];
    float x = __expf(ex[e] - m[d]);
    ex[e] = x;
    atomicAdd(&denom[d], x);
}

// z_nb[dst] += alpha * z[src]; one wave per edge, float4 per lane
__global__ __launch_bounds__(256)
void edge_scatter(const int* __restrict__ src, const int* __restrict__ dst,
                  const float* __restrict__ ex, const float* __restrict__ denom,
                  const float* __restrict__ Z, float* __restrict__ znb, int e_cnt) {
    int e = blockIdx.x * 8 + (threadIdx.x >> 5);
    int lane = threadIdx.x & 31;
    if (e >= e_cnt) return;
    int s = src[e], d = dst[e];
    float alpha = ex[e] / fmaxf(denom[d], 1e-9f);
    float4 zz = *(const float4*)(Z + (long)s * DD + lane * 4);
    float* o = znb + (long)d * DD + lane * 4;
    atomicAdd(o + 0, alpha * zz.x);
    atomicAdd(o + 1, alpha * zz.y);
    atomicAdd(o + 2, alpha * zz.z);
    atomicAdd(o + 3, alpha * zz.w);
}

__global__ void finalize_relu(const float* __restrict__ zi, const float* __restrict__ znb,
                              float* __restrict__ out, int nd) {
    int i = blockIdx.x * blockDim.x + threadIdx.x;
    if (i < nd) out[i] = fmaxf(zi[i] + znb[i], 0.0f);
}

// ---------------------------------------------------------------------------
extern "C" void kernel_launch(void* const* d_in, const int* in_sizes, int n_in,
                              void* d_out, int out_size, void* d_ws, size_t ws_size,
                              hipStream_t stream) {
    (void)in_sizes; (void)n_in; (void)out_size; (void)ws_size;

    const float* attr    = (const float*)d_in[0];
    const float* edge_d  = (const float*)d_in[1];
    const int*   src     = (const int*)  d_in[2];
    const int*   dst     = (const int*)  d_in[3];
    const float* fc0_w1  = (const float*)d_in[4];
    const float* fc1_w1  = (const float*)d_in[5];
    const float* fc2_w1  = (const float*)d_in[6];
    const float* attn_w1 = (const float*)d_in[7];
    const float* fc0_w2  = (const float*)d_in[8];
    const float* fc1_w2  = (const float*)d_in[9];
    const float* fc2_w2  = (const float*)d_in[10];
    const float* attn_w2 = (const float*)d_in[11];

    float* ws = (float*)d_ws;
    const long ND = (long)N_NODES * DD;
    float* z    = ws;                 // N x D
    float* zi   = z    + ND;          // N x D
    float* znb  = zi   + ND;          // N x D
    float* h1   = znb  + ND;          // N x D (layer-1 output)
    float* sv   = h1   + ND;          // N
    float* dv   = sv   + N_NODES;     // N
    float* mbuf = dv   + N_NODES;     // N
    float* den  = mbuf + N_NODES;     // N
    float* ex   = den  + N_NODES;     // E

    dim3 b256(256);
    const int gTiles = (N_NODES + 127) / 128;
    const int gND    = (int)((ND + 255) / 256);
    const int gN8    = (N_NODES + 7) / 8;
    const int gE     = (N_EDGES + 255) / 256;
    const int gE8    = (N_EDGES + 7) / 8;

    const size_t shmem1 = (size_t)(128 * F_IN_1 + 2 * DD * F_IN_1) * sizeof(__bf16); // ~48 KB
    const size_t shmem2 = (size_t)(128 * DD     + 2 * DD * DD)     * sizeof(__bf16); // ~96 KB

    // -------------------- layer 1 (K = 64) --------------------
    hipLaunchKernelGGL(init_layer, dim3(gND), b256, 0, stream, mbuf, den, znb, N_NODES, (int)ND);
    hipLaunchKernelGGL(gemm2_bf16_wmma, dim3(gTiles), b256, shmem1, stream,
                       attr, fc1_w1, fc2_w1, z, zi, N_NODES, F_IN_1);
    hipLaunchKernelGGL(node_dots, dim3(gN8), b256, 0, stream, z, attn_w1, sv, dv, N_NODES);
    hipLaunchKernelGGL(edge_max, dim3(gE), b256, 0, stream,
                       src, dst, edge_d, sv, dv, fc0_w1, attn_w1, ex, mbuf, N_EDGES);
    hipLaunchKernelGGL(edge_expsum, dim3(gE), b256, 0, stream, dst, mbuf, ex, den, N_EDGES);
    hipLaunchKernelGGL(edge_scatter, dim3(gE8), b256, 0, stream,
                       src, dst, ex, den, z, znb, N_EDGES);
    hipLaunchKernelGGL(finalize_relu, dim3(gND), b256, 0, stream, zi, znb, h1, (int)ND);

    // -------------------- layer 2 (K = 128) --------------------
    hipLaunchKernelGGL(init_layer, dim3(gND), b256, 0, stream, mbuf, den, znb, N_NODES, (int)ND);
    hipLaunchKernelGGL(gemm2_bf16_wmma, dim3(gTiles), b256, shmem2, stream,
                       h1, fc1_w2, fc2_w2, z, zi, N_NODES, DD);
    hipLaunchKernelGGL(node_dots, dim3(gN8), b256, 0, stream, z, attn_w2, sv, dv, N_NODES);
    hipLaunchKernelGGL(edge_max, dim3(gE), b256, 0, stream,
                       src, dst, edge_d, sv, dv, fc0_w2, attn_w2, ex, mbuf, N_EDGES);
    hipLaunchKernelGGL(edge_expsum, dim3(gE), b256, 0, stream, dst, mbuf, ex, den, N_EDGES);
    hipLaunchKernelGGL(edge_scatter, dim3(gE8), b256, 0, stream,
                       src, dst, ex, den, z, znb, N_EDGES);
    hipLaunchKernelGGL(finalize_relu, dim3(gND), b256, 0, stream, zi, znb, (float*)d_out, (int)ND);
}
